// Generator_9715216024086
// MI455X (gfx1250) — matrix operand order repaired and checked by reading, not compiled, not run on hardware
//
#include <hip/hip_runtime.h>
#include <hip/hip_bf16.h>

#define B_   8
#define H_   512
#define W_   512
#define N_   (H_ * W_)      // 262144 = 2^18
#define S_   1024
#define ZD_  128
#define C_   256
#define KIN_ (3 + ZD_)      // 131

#define SC_IN_F 0.123560424f   // sqrt(2/131)
#define SC_C_F  0.088388348f   // sqrt(2/256)
#define SC_2C_F 0.0625f        // sqrt(2/512)

#define KCH  32                // K-chunk per WMMA step
#define BROW 40                // padded LDS row for B^T tile (16B aligned, spreads banks)

typedef __attribute__((ext_vector_type(16))) _Float16 v16h;
typedef __attribute__((ext_vector_type(8)))  _Float16 v8h;
typedef __attribute__((ext_vector_type(4)))  _Float16 v4h;
typedef __attribute__((ext_vector_type(8)))  float    v8f;
typedef __attribute__((ext_vector_type(4)))  unsigned v4u;
typedef __attribute__((ext_vector_type(8)))  int      v8i_;
typedef __attribute__((ext_vector_type(4)))  int      v4i_;

// ---------------------------------------------------------------------------
// WMMA fragment loaders (wave32, 16x16x32 f16 layouts per CDNA5 ISA 7.12.2)
// A (16x32, MxK): lane L holds row m=L&15; kh=L>>4; element e: k = (e>>3)*16 + kh*8 + (e&7)
// B (32x16, KxN): lane L holds col n=L&15; kh=L>>4; element e: k = (e>>3)*16 + kh*8 + (e&7)
// ---------------------------------------------------------------------------
__device__ inline v16h frag_a_lds(const _Float16* ldsA, int rowBase, int lane) {
  const int m = lane & 15, kh = lane >> 4;
  const _Float16* p = ldsA + (rowBase + m) * KCH + kh * 8;
  v8h g0 = *(const v8h*)p;          // k = kh*8 + 0..7
  v8h g1 = *(const v8h*)(p + 16);   // k = 16 + kh*8 + 0..7
  v16h a;
#pragma unroll
  for (int e = 0; e < 8; ++e) { a[e] = g0[e]; a[8 + e] = g1[e]; }
  return a;
}

__device__ inline v16h frag_b_lds(const _Float16* ldsBt, int colBase, int lane) {
  const int n = lane & 15, kh = lane >> 4;
  const _Float16* p = ldsBt + (colBase + n) * BROW + kh * 8;
  v8h g0 = *(const v8h*)p;          // k = kh*8 + 0..7
  v8h g1 = *(const v8h*)(p + 16);   // k = 16 + kh*8 + 0..7
  v16h b;
#pragma unroll
  for (int e = 0; e < 8; ++e) { b[e] = g0[e]; b[8 + e] = g1[e]; }
  return b;
}

// ---------------------------------------------------------------------------
// 1) Segment accumulation: sums(B,S,3), counts(B,S) via global f32 atomics
// ---------------------------------------------------------------------------
__global__ __launch_bounds__(256) void k_seg_accum(const float* __restrict__ img,
                                                   const int* __restrict__ seg,
                                                   float* __restrict__ sums,
                                                   float* __restrict__ counts) {
  const int g = blockIdx.x * 256 + threadIdx.x;   // over B*N
  const int b = g >> 18;
  const int p = g & (N_ - 1);
  const int s = seg[g];
  const int base = (b * S_ + s) * 3;
  atomicAdd(&sums[base + 0], img[((size_t)(b * 3 + 0) << 18) + p]);
  atomicAdd(&sums[base + 1], img[((size_t)(b * 3 + 1) << 18) + p]);
  atomicAdd(&sums[base + 2], img[((size_t)(b * 3 + 2) << 18) + p]);
  atomicAdd(&counts[b * S_ + s], 1.0f);
}

// ---------------------------------------------------------------------------
// 2) feats = sums/(counts+eps); cat = [feats, z]*SC_IN; x = cat @ proj_w^T + b
// ---------------------------------------------------------------------------
__global__ __launch_bounds__(256) void k_feats_proj(const float* __restrict__ sums,
                                                    const float* __restrict__ counts,
                                                    const float* __restrict__ z,
                                                    const float* __restrict__ pw,
                                                    const float* __restrict__ pb,
                                                    float* __restrict__ x) {
  __shared__ float cat[KIN_];
  const int bs = blockIdx.x;           // b*S + s
  const int b  = bs >> 10;
  const int t  = threadIdx.x;
  if (t < 3) cat[t] = sums[bs * 3 + t] / (counts[bs] + 1e-6f) * SC_IN_F;
  else if (t < KIN_) cat[t] = z[b * ZD_ + (t - 3)] * SC_IN_F;
  __syncthreads();
  float acc = pb[t];
#pragma unroll 1
  for (int j = 0; j < KIN_; ++j) acc += cat[j] * pw[t * KIN_ + j];
  x[(size_t)bs * C_ + t] = acc;
}

// ---------------------------------------------------------------------------
// 3) One-time f32 -> f16 convert (for A: 8M elems, L2-resident at 16MB)
// ---------------------------------------------------------------------------
__global__ __launch_bounds__(256) void k_f32_to_f16(const float* __restrict__ in,
                                                    _Float16* __restrict__ out) {
  const size_t g = (size_t)blockIdx.x * 256 + threadIdx.x;   // 4 elems / thread
  const float4 v = ((const float4*)in)[g];
  v4h o;
  o[0] = (_Float16)v.x; o[1] = (_Float16)v.y; o[2] = (_Float16)v.z; o[3] = (_Float16)v.w;
  ((v4h*)out)[g] = o;
}

// ---------------------------------------------------------------------------
// 4a) h = (x * SC_C) @ W^T + bias, written TRANSPOSED: h16t[b][c][s] (C x S).
//     WG = 256 threads = 8 waves -> 4x2 tiles of 16x16 -> 64x32 output block.
// ---------------------------------------------------------------------------
__global__ __launch_bounds__(256) void k_gemm_proj(const float* __restrict__ x,
                                                   const float* __restrict__ Wt,   // (C,C) row-major
                                                   const float* __restrict__ bias,
                                                   _Float16* __restrict__ h16t) {
  __shared__ alignas(16) _Float16 ldsA[64 * KCH];
  __shared__ alignas(16) _Float16 ldsBt[32 * BROW];
  const int n0 = blockIdx.x * 32;
  const int m0 = blockIdx.y * 64;
  const int b  = blockIdx.z;
  const int tid = threadIdx.x, lane = tid & 31, wv = tid >> 5;
  const int tm = wv >> 1, tn = wv & 1;
  const float* xb = x + (size_t)b * S_ * C_;
  v8f acc = {};
  for (int kc = 0; kc < C_; kc += KCH) {
    {                                   // stage A: 8 contiguous K per thread -> 1 ds_store_b128
      const int m = tid >> 2, k8 = (tid & 3) * 8;
      const float4 f0 = *(const float4*)&xb[(size_t)(m0 + m) * C_ + kc + k8];
      const float4 f1 = *(const float4*)&xb[(size_t)(m0 + m) * C_ + kc + k8 + 4];
      v8h o;
      o[0] = (_Float16)(f0.x * SC_C_F); o[1] = (_Float16)(f0.y * SC_C_F);
      o[2] = (_Float16)(f0.z * SC_C_F); o[3] = (_Float16)(f0.w * SC_C_F);
      o[4] = (_Float16)(f1.x * SC_C_F); o[5] = (_Float16)(f1.y * SC_C_F);
      o[6] = (_Float16)(f1.z * SC_C_F); o[7] = (_Float16)(f1.w * SC_C_F);
      *(v8h*)&ldsA[m * KCH + k8] = o;
    }
    if (tid < 128) {                    // stage B^T: ldsBt[n][k] = W[n0+n][kc+k]
      const int n = tid >> 2, k8 = (tid & 3) * 8;
      const float4 f0 = *(const float4*)&Wt[(size_t)(n0 + n) * C_ + kc + k8];
      const float4 f1 = *(const float4*)&Wt[(size_t)(n0 + n) * C_ + kc + k8 + 4];
      v8h o;
      o[0] = (_Float16)f0.x; o[1] = (_Float16)f0.y; o[2] = (_Float16)f0.z; o[3] = (_Float16)f0.w;
      o[4] = (_Float16)f1.x; o[5] = (_Float16)f1.y; o[6] = (_Float16)f1.z; o[7] = (_Float16)f1.w;
      *(v8h*)&ldsBt[n * BROW + k8] = o;
    }
    __syncthreads();
    const v16h a  = frag_a_lds(ldsA, tm * 16, lane);
    const v16h bf = frag_b_lds(ldsBt, tn * 16, lane);
    acc = __builtin_amdgcn_wmma_f32_16x16x32_f16(false, a, false, bf, (short)0, acc, false, false);
    __syncthreads();
  }
  // epilogue: one b128 store per lane into transposed layout h16t[c][s]
  const int col   = n0 + tn * 16 + (lane & 15);
  const int rbase = m0 + tm * 16 + (lane >> 4) * 8;
  const float bc = bias[col];
  v8h o;
#pragma unroll
  for (int v = 0; v < 8; ++v) o[v] = (_Float16)(acc[v] + bc);
  *(v8h*)&h16t[((size_t)b * C_ + col) * S_ + rbase] = o;
}

// ---------------------------------------------------------------------------
// 4b) hA = A @ h   (per batch; M=S, N=C, K=S), f16 in, f32 out.
//     A tile (64x32 f16) moved by the Tensor Data Mover (wave 0 issues,
//     TENSORcnt-synchronized); B tile staged from transposed h16t with
//     contiguous b128 copies.
// ---------------------------------------------------------------------------
__global__ __launch_bounds__(256) void k_gemm_A_h(const _Float16* __restrict__ A16,
                                                  const _Float16* __restrict__ h16t,
                                                  float* __restrict__ out) {
  __shared__ alignas(16) _Float16 ldsA[64 * KCH];
  __shared__ alignas(16) _Float16 ldsBt[32 * BROW];
  const int n0 = blockIdx.x * 32;
  const int m0 = blockIdx.y * 64;
  const int b  = blockIdx.z;
  const int tid = threadIdx.x, lane = tid & 31, wv = tid >> 5;
  const int tm = wv >> 1, tn = wv & 1;
  const _Float16* Ab  = A16  + (size_t)b * S_ * S_;
  const _Float16* hbt = h16t + (size_t)b * C_ * S_;

  // TDM descriptor group 1 (loop-invariant): 8-byte elements.
  // tile = 8 x 64 elements (64B rows), tensor row = 256 elements, stride = 256.
  v8i_ g1;
  g1[0] = (int)(3u << 16);                               // data_size = 8B
  g1[1] = (int)((256u & 0xffffu) << 16);                 // tensor_dim0 = 256
  g1[2] = (int)((1024u & 0xffffu) << 16);                // tensor_dim1 = 1024
  g1[3] = (int)(8u << 16);                               // tile_dim0 = 8
  g1[4] = (int)64u;                                      // tile_dim1 = 64, tile_dim2 = 0
  g1[5] = (int)256u;                                     // tensor_dim0_stride = 256
  g1[6] = 0;
  g1[7] = 0;
  const v4i_ gz4 = {};
  const v8i_ gz8 = {};
  const unsigned ldsa_off = (unsigned)(size_t)(void*)&ldsA[0];

  v8f acc = {};
  for (int kc = 0; kc < S_; kc += KCH) {
    if (tid < 32) {                      // wave 0: issue TDM load of A tile -> LDS
      const unsigned long long ga =
          (unsigned long long)(const void*)(Ab + (size_t)m0 * S_ + kc);
      v4u g0;
      g0[0] = 1u;                                        // count = 1
      g0[1] = ldsa_off;                                  // lds_addr
      g0[2] = (unsigned)ga;                              // global_addr[31:0]
      g0[3] = (unsigned)((ga >> 32) & 0x1FFFFFFu) | (2u << 30);  // addr[56:32] | type=2
      __builtin_amdgcn_tensor_load_to_lds(g0, g1, gz4, gz4, gz8, 0);
      __builtin_amdgcn_s_wait_tensorcnt(0);
    }
    if (tid < 128) {                     // stage B^T tile: contiguous v8h copies
      const int n = tid >> 2, k8 = (tid & 3) * 8;
      *(v8h*)&ldsBt[n * BROW + k8] =
          *(const v8h*)&hbt[(size_t)(n0 + n) * S_ + kc + k8];
      if (kc + KCH < S_)
        __builtin_prefetch(&hbt[(size_t)(n0 + n) * S_ + kc + KCH + k8], 0, 3);
    }
    __syncthreads();
    const v16h a  = frag_a_lds(ldsA, tm * 16, lane);
    const v16h bf = frag_b_lds(ldsBt, tn * 16, lane);
    acc = __builtin_amdgcn_wmma_f32_16x16x32_f16(false, a, false, bf, (short)0, acc, false, false);
    __syncthreads();
  }
  const int col   = n0 + tn * 16 + (lane & 15);
  const int rbase = m0 + tm * 16 + (lane >> 4) * 8;
#pragma unroll
  for (int v = 0; v < 8; ++v)
    out[((size_t)b * S_ + rbase + v) * C_ + col] = acc[v];
}

// ---------------------------------------------------------------------------
// 4c) m = mean over S of hA   (one WG per batch; thread = channel; coalesced)
// ---------------------------------------------------------------------------
__global__ __launch_bounds__(256) void k_colmean(const float* __restrict__ hA,
                                                 float* __restrict__ m) {
  const int b = blockIdx.x, c = threadIdx.x;
  float s = 0.0f;
  for (int r = 0; r < S_; ++r) s += hA[((size_t)b * S_ + r) * C_ + c];
  m[b * C_ + c] = s * (1.0f / S_);
}

// ---------------------------------------------------------------------------
// 4d) AdaIN MLP: s1 = leaky(m*SC_C @ w1^T + b1); style = s1*SC_2C @ w2^T + b2
// ---------------------------------------------------------------------------
__global__ __launch_bounds__(512) void k_ada(const float* __restrict__ m,
                                             const float* __restrict__ w1,
                                             const float* __restrict__ b1,
                                             const float* __restrict__ w2,
                                             const float* __restrict__ b2,
                                             float* __restrict__ style) {
  __shared__ float mlds[C_];
  __shared__ float s1[2 * C_];
  const int b = blockIdx.x, t = threadIdx.x;
  if (t < C_) mlds[t] = m[b * C_ + t] * SC_C_F;
  __syncthreads();
  float a1 = b1[t];
  for (int j = 0; j < C_; ++j) a1 += mlds[j] * w1[(size_t)t * C_ + j];
  a1 = (a1 > 0.0f) ? a1 : 0.2f * a1;
  s1[t] = a1 * SC_2C_F;
  __syncthreads();
  float a2 = b2[t];
  for (int j = 0; j < 2 * C_; ++j) a2 += s1[j] * w2[(size_t)t * (2 * C_) + j];
  style[b * (2 * C_) + t] = a2;
}

// ---------------------------------------------------------------------------
// 4e) x = leaky(gamma*hA + beta)
// ---------------------------------------------------------------------------
__global__ __launch_bounds__(256) void k_modulate(const float* __restrict__ hA,
                                                  const float* __restrict__ style,
                                                  float* __restrict__ x) {
  const int g = blockIdx.x * 256 + threadIdx.x;   // over B*S*C
  const int b = g / (S_ * C_);
  const int c = g & (C_ - 1);
  const float gam = style[b * (2 * C_) + c];
  const float bet = style[b * (2 * C_) + C_ + c];
  float v = gam * hA[g] + bet;
  x[g] = (v > 0.0f) ? v : 0.2f * v;
}

// ---------------------------------------------------------------------------
// 5) color = x*SC_C @ rgb_w^T + rgb_b ; then gather + tanh
// ---------------------------------------------------------------------------
__global__ __launch_bounds__(256) void k_rgb(const float* __restrict__ x,
                                             const float* __restrict__ rw,
                                             const float* __restrict__ rb,
                                             float* __restrict__ color) {
  const int g = blockIdx.x * 256 + threadIdx.x;   // over B*S*3
  if (g >= B_ * S_ * 3) return;
  const int ch = g % 3, bs = g / 3;
  const float* xr = x + (size_t)bs * C_;
  const float* wr = rw + ch * C_;
  float d = 0.0f;
  for (int j = 0; j < C_; ++j) d += xr[j] * wr[j];
  color[g] = d * SC_C_F + rb[ch];
}

__global__ __launch_bounds__(256) void k_gather(const int* __restrict__ seg,
                                                const float* __restrict__ color,
                                                float* __restrict__ out) {
  const int g = blockIdx.x * 256 + threadIdx.x;   // over B*N
  const int b = g >> 18;
  const int p = g & (N_ - 1);
  const int s = seg[g];
  const float* c = color + ((size_t)b * S_ + s) * 3;
  out[((size_t)(b * 3 + 0) << 18) + p] = tanhf(c[0]);
  out[((size_t)(b * 3 + 1) << 18) + p] = tanhf(c[1]);
  out[((size_t)(b * 3 + 2) << 18) + p] = tanhf(c[2]);
}

// ---------------------------------------------------------------------------
extern "C" void kernel_launch(void* const* d_in, const int* in_sizes, int n_in,
                              void* d_out, int out_size, void* d_ws, size_t ws_size,
                              hipStream_t stream) {
  (void)in_sizes; (void)n_in; (void)out_size; (void)ws_size;
  const float* z   = (const float*)d_in[0];
  const float* img = (const float*)d_in[1];
  const int*   seg = (const int*)d_in[2];
  const float* A   = (const float*)d_in[3];
  const float* pw  = (const float*)d_in[4];
  const float* pb  = (const float*)d_in[5];
  const float* bw  = (const float*)d_in[6];
  const float* bb  = (const float*)d_in[7];
  const float* aw1 = (const float*)d_in[8];
  const float* ab1 = (const float*)d_in[9];
  const float* aw2 = (const float*)d_in[10];
  const float* ab2 = (const float*)d_in[11];
  const float* rw  = (const float*)d_in[12];
  const float* rb  = (const float*)d_in[13];
  float* out = (float*)d_out;

  char* ws = (char*)d_ws;
  size_t off = 0;
  auto alloc = [&](size_t bytes) -> void* {
    void* p = ws + off;
    off = (off + bytes + 255) & ~(size_t)255;
    return p;
  };
  float*    sums   = (float*)   alloc((size_t)B_ * S_ * 3 * 4);
  float*    counts = (float*)   alloc((size_t)B_ * S_ * 4);
  float*    x      = (float*)   alloc((size_t)B_ * S_ * C_ * 4);
  _Float16* h16t   = (_Float16*)alloc((size_t)B_ * C_ * S_ * 2);
  _Float16* A16    = (_Float16*)alloc((size_t)B_ * S_ * S_ * 2);
  float*    hA     = (float*)   alloc((size_t)B_ * S_ * C_ * 4);
  float*    mvec   = (float*)   alloc((size_t)B_ * C_ * 4);
  float*    style  = (float*)   alloc((size_t)B_ * 2 * C_ * 4);
  float*    color  = (float*)   alloc((size_t)B_ * S_ * 3 * 4);

  (void)hipMemsetAsync(sums,   0, (size_t)B_ * S_ * 3 * 4, stream);
  (void)hipMemsetAsync(counts, 0, (size_t)B_ * S_ * 4,     stream);

  k_seg_accum<<<(B_ * N_) / 256, 256, 0, stream>>>(img, seg, sums, counts);
  k_feats_proj<<<B_ * S_, 256, 0, stream>>>(sums, counts, z, pw, pb, x);
  k_f32_to_f16<<<(B_ * S_ * S_) / (4 * 256), 256, 0, stream>>>(A, A16);

  const dim3 gg(C_ / 32, S_ / 64, B_);
  for (int i = 0; i < 8; ++i) {
    k_gemm_proj<<<gg, 256, 0, stream>>>(x, bw + (size_t)i * C_ * C_, bb + i * C_, h16t);
    k_gemm_A_h <<<gg, 256, 0, stream>>>(A16, h16t, hA);
    k_colmean  <<<B_, 256, 0, stream>>>(hA, mvec);
    k_ada      <<<B_, 512, 0, stream>>>(mvec,
                                        aw1 + (size_t)i * 2 * C_ * C_, ab1 + i * 2 * C_,
                                        aw2 + (size_t)i * 2 * C_ * 2 * C_, ab2 + i * 2 * C_,
                                        style);
    k_modulate <<<(B_ * S_ * C_) / 256, 256, 0, stream>>>(hA, style, x);
  }

  k_rgb   <<<(B_ * S_ * 3 + 255) / 256, 256, 0, stream>>>(x, rw, rb, color);
  k_gather<<<(B_ * N_) / 256, 256, 0, stream>>>(seg, color, out);
}
